// RopelessMLA_90305982366258
// MI455X (gfx1250) — matrix operand order repaired and checked by reading, not compiled, not run on hardware
//
#include <hip/hip_runtime.h>

// ---------------- problem constants (match reference) ----------------
#define B_   2
#define S_   2048
#define D_   1024
#define H_   16
#define L_   512
#define DH_  64
#define BS_  (B_ * S_)          // 4096 total rows
#define LN_EPS 1e-5f

typedef __attribute__((ext_vector_type(16))) __bf16 v16bf;
typedef __attribute__((ext_vector_type(8)))  __bf16 v8bf;   // 16 bytes
typedef __attribute__((ext_vector_type(8)))  float  v8f;
typedef __attribute__((ext_vector_type(4)))  int    v4i;

// optional CDNA5 async global->LDS path (probe-guarded; falls back to b128 copies)
#if __has_builtin(__builtin_amdgcn_global_load_async_to_lds_b128) && \
    __has_builtin(__builtin_amdgcn_s_wait_asynccnt)
#define USE_ASYNC_LDS 1
#else
#define USE_ASYNC_LDS 0
#endif

#if USE_ASYNC_LDS
typedef __attribute__((address_space(1))) v4i* gptr_v4i;   // global
typedef __attribute__((address_space(3))) v4i* lptr_v4i;   // LDS
__device__ __forceinline__ void g2l_b128(const __bf16* g, __bf16* l) {
  __builtin_amdgcn_global_load_async_to_lds_b128(
      (gptr_v4i)(g), (lptr_v4i)(l), 0, 0);
}
#endif

// ---------------- WMMA fragment helpers (CDNA5 wave32 layouts) ----------------
// A fragment 16x32 bf16 from a row-major LDS tile [m][k]:
// lane (m = lane&15, kh = (lane>>4)*8) holds k = kh..kh+7 and 16+kh..+7,
// i.e. two contiguous 16-byte runs -> two ds_load_b128.
__device__ __forceinline__ v16bf load_frag_a_rm(const __bf16* base, int strideBf) {
  const int lane = threadIdx.x & 31;
  const int m  = lane & 15;
  const int kh = (lane >> 4) << 3;   // 0 or 8
  const v8bf* p = (const v8bf*)(base + m * strideBf + kh);
  const v8bf lo = p[0];              // k = kh .. kh+7
  const v8bf hi = p[2];              // k = 16+kh .. 16+kh+7
  return __builtin_shufflevector(lo, hi, 0, 1, 2, 3, 4, 5, 6, 7,
                                         8, 9, 10, 11, 12, 13, 14, 15);
}

// B fragment 32x16 bf16 from a TRANSPOSED LDS tile [n][k]:
// lane (n = lane&15, kh = (lane>>4)*8) holds k = kh..kh+7 and 16+kh..+7 (same order as A).
__device__ __forceinline__ v16bf load_frag_b_t(const __bf16* base, int strideBf) {
  const int lane = threadIdx.x & 31;
  const int n  = lane & 15;
  const int kh = (lane >> 4) << 3;   // 0 or 8
  const v8bf* p = (const v8bf*)(base + n * strideBf + kh);
  const v8bf lo = p[0];              // k = kh .. kh+7
  const v8bf hi = p[2];              // k = 16+kh .. 16+kh+7
  return __builtin_shufflevector(lo, hi, 0, 1, 2, 3, 4, 5, 6, 7,
                                         8, 9, 10, 11, 12, 13, 14, 15);
}

__device__ __forceinline__ v8f wmma_bf16(v16bf a, v16bf b, v8f c) {
  return __builtin_amdgcn_wmma_f32_16x16x32_bf16(
      /*neg_a=*/false, a, /*neg_b=*/false, b,
      /*c_mod=*/(short)0, c, /*reuse_a=*/false, /*reuse_b=*/false);
}

// C/D element mapping: VGPR v holds (m = v + 8*(lane>=16), n = lane&15).

// convert 8 consecutive f32 to one 16-byte bf16 vector (v_cvt_pk_bf16_f32 x4)
__device__ __forceinline__ v8bf cvt8(const float* s) {
  v8bf r;
#pragma unroll
  for (int i = 0; i < 8; ++i) r[i] = (__bf16)s[i];
  return r;
}

// ---------------- generic f32-in, WMMA-bf16 GEMM ----------------
// C[M,N] = A[M,K] * op(B)[K,N]; A,B f32 in global, staged to LDS as bf16
// (B staged transposed: BsT[n][k]). Block = 128 thr (4 waves), tile 128x64,
// K-step 32. Each wave owns a 32x64 stripe: 2 A-frags x 4 B-frags = 8 WMMA
// per k-step, with B-fragments reused from registers across the two row blocks.
#define TM 128
#define TN 64
#define TK 32

__global__ __launch_bounds__(128)
void gemm_bf16_wmma(const float* __restrict__ A, int lda, long sAz,
                    const float* __restrict__ B, int ldb, long sBz, int transB,
                    float* __restrict__ Cf, __bf16* __restrict__ Cbf,
                    int ldc, long sCz, int transC, int M, int N, int K)
{
  __shared__ __attribute__((aligned(16))) __bf16 As [TM][TK + 8];  // [m][k], 80B rows
  __shared__ __attribute__((aligned(16))) __bf16 BsT[TN][TK + 8];  // [n][k]

  A += (long)blockIdx.z * sAz;
  B += (long)blockIdx.z * sBz;
  const long cofs = (long)blockIdx.z * sCz;

  const int m0 = blockIdx.y * TM;
  const int n0 = blockIdx.x * TN;
  const int tid  = threadIdx.x;
  const int wave = tid >> 5;
  const int lane = tid & 31;

  v8f acc[2][4];
#pragma unroll
  for (int mi = 0; mi < 2; ++mi)
#pragma unroll
    for (int t = 0; t < 4; ++t)
#pragma unroll
      for (int v = 0; v < 8; ++v) acc[mi][t][v] = 0.f;

  for (int k0 = 0; k0 < K; k0 += TK) {
    // stage A tile 128x32: each thread stages one full 32-float row
    {
      const int r = tid;
      const float* src = A + (long)(m0 + r) * lda + k0;
      if (k0 + TK < K) __builtin_prefetch(src + TK, 0, 1);   // global_prefetch_b8
      v8bf* dst = (v8bf*)&As[r][0];
      dst[0] = cvt8(src);
      dst[1] = cvt8(src + 8);
      dst[2] = cvt8(src + 16);
      dst[3] = cvt8(src + 24);
    }
    // stage op(B) tile transposed: BsT[n][k]
    {
      const int n  = tid >> 1;          // local n 0..63
      const int kh = (tid & 1) * 16;    // local k 0 or 16
      float tv[16];
      if (transB) {                      // stored [N,K]: contiguous over k
        const float* src = B + (long)(n0 + n) * ldb + k0 + kh;
#pragma unroll
        for (int i = 0; i < 16; ++i) tv[i] = src[i];
      } else {                           // stored [K,N]: strided column read
#pragma unroll
        for (int i = 0; i < 16; ++i) tv[i] = B[(long)(k0 + kh + i) * ldb + n0 + n];
      }
      v8bf* dst = (v8bf*)&BsT[n][kh];
      dst[0] = cvt8(tv);
      dst[1] = cvt8(tv + 8);
    }
    __syncthreads();

    const v16bf a0 = load_frag_a_rm(&As[wave * 32][0],      TK + 8);
    const v16bf a1 = load_frag_a_rm(&As[wave * 32 + 16][0], TK + 8);
#pragma unroll
    for (int t = 0; t < 4; ++t) {
      const v16bf bfrag = load_frag_b_t(&BsT[t * 16][0], TK + 8);
      acc[0][t] = wmma_bf16(a0, bfrag, acc[0][t]);
      acc[1][t] = wmma_bf16(a1, bfrag, acc[1][t]);
    }
    __syncthreads();
  }

  const int nloc = lane & 15;
  const int mb   = (lane >> 4) << 3;
#pragma unroll
  for (int mi = 0; mi < 2; ++mi) {
#pragma unroll
    for (int t = 0; t < 4; ++t) {
#pragma unroll
      for (int v = 0; v < 8; ++v) {
        const int m = m0 + wave * 32 + mi * 16 + mb + v;
        const int n = n0 + t * 16 + nloc;
        if (m < M && n < N) {
          const float val = acc[mi][t][v];
          const long idx = cofs + (transC ? ((long)n * ldc + m) : ((long)m * ldc + n));
          if (Cf)  Cf[idx]  = val;
          if (Cbf) Cbf[idx] = (__bf16)val;
        }
      }
    }
  }
}

// ---------------- LayerNorm over L, emits f32 (to d_out) + bf16 (to ws) ----------------
__global__ __launch_bounds__(256)
void layernorm_kernel(const float* __restrict__ c,
                      const float* __restrict__ gamma,
                      const float* __restrict__ beta,
                      float* __restrict__ outf,
                      __bf16* __restrict__ outbf)
{
  const int row = blockIdx.x;
  const float* cr = c + (long)row * L_;
  float s = 0.f, s2 = 0.f;
  for (int i = threadIdx.x; i < L_; i += 256) {
    const float v = cr[i];
    s += v; s2 += v * v;
  }
#pragma unroll
  for (int d = 1; d < 32; d <<= 1) {
    s  += __shfl_xor(s,  d, 32);
    s2 += __shfl_xor(s2, d, 32);
  }
  __shared__ float shs[8], shs2[8];
  const int wave = threadIdx.x >> 5, lane = threadIdx.x & 31;
  if (lane == 0) { shs[wave] = s; shs2[wave] = s2; }
  __syncthreads();
  if (wave == 0) {
    s  = (lane < 8) ? shs[lane]  : 0.f;
    s2 = (lane < 8) ? shs2[lane] : 0.f;
#pragma unroll
    for (int d = 1; d < 8; d <<= 1) {
      s  += __shfl_xor(s,  d, 32);
      s2 += __shfl_xor(s2, d, 32);
    }
    if (lane == 0) { shs[0] = s; shs2[0] = s2; }
  }
  __syncthreads();
  const float mean = shs[0] * (1.f / L_);
  const float var  = shs2[0] * (1.f / L_) - mean * mean;
  const float rstd = rsqrtf(var + LN_EPS);
  for (int i = threadIdx.x; i < L_; i += 256) {
    const float y = (cr[i] - mean) * rstd * gamma[i] + beta[i];
    outf[(long)row * L_ + i]  = y;
    outbf[(long)row * L_ + i] = (__bf16)y;
  }
}

// ---------------- flash attention over the latent space ----------------
// scores[s,t] = (tmp[s,:] . c_kv[t,:]) / 8 with K = L = 512 (bf16 WMMA),
// online causal softmax, ctx += P @ V. V is pre-transposed to [D][BS] so its
// LDS tile is staged directly in [n][k] fragment order. bf16 tiles are staged
// via the CDNA5 async global->LDS path when available.
__global__ __launch_bounds__(128)
void mla_flash_kernel(const __bf16* __restrict__ qabs,   // [H][BS][L] bf16
                      const __bf16* __restrict__ ckv,    // [BS][L]    bf16
                      const __bf16* __restrict__ vmat_t, // [D][BS]    bf16
                      float* __restrict__ ctx)           // [BS][D]    f32
{
  __shared__ __attribute__((aligned(16))) __bf16 Qs [64][72];  // [q][l]   144B rows
  __shared__ __attribute__((aligned(16))) __bf16 Ks [64][72];  // [key][l] == B [n][k]
  __shared__ __attribute__((aligned(16))) __bf16 VsT[64][72];  // [dh][key] == B [n][k]
  __shared__ __attribute__((aligned(16))) __bf16 Ps [64][72];  // [q][key] == A [m][k]

  const int qt  = blockIdx.x;
  const int h   = blockIdx.y;
  const int b   = blockIdx.z;
  const int tid  = threadIdx.x;
  const int wave = tid >> 5;
  const int lane = tid & 31;
  const int mb   = (lane >> 4) << 3;   // 0 or 8
  const int nloc = lane & 15;

  const long rowbase = (long)b * S_;
  const int  qr0 = qt * 64;
  const __bf16* Q = qabs + (long)h * BS_ * L_ + (rowbase + qr0) * L_;

  v8f   cacc[4];
  float mrow[8], lrow[8];
#pragma unroll
  for (int t = 0; t < 4; ++t)
#pragma unroll
    for (int v = 0; v < 8; ++v) cacc[t][v] = 0.f;
#pragma unroll
  for (int v = 0; v < 8; ++v) { mrow[v] = -3.0e38f; lrow[v] = 0.f; }

  for (int j = 0; j <= qt; ++j) {           // causal: key tiles up to the diagonal
    v8f sacc[4];
#pragma unroll
    for (int t = 0; t < 4; ++t)
#pragma unroll
      for (int v = 0; v < 8; ++v) sacc[t][v] = 0.f;

    // ---- S tile = Q[64,512] @ Ckv[j*64..+64, 512]^T, chunked over L ----
    for (int l0 = 0; l0 < L_; l0 += 64) {
      const int r = tid >> 1;
      const int c = (tid & 1) * 32;
      const __bf16* qsrc = Q + (long)r * L_ + l0 + c;
      const __bf16* ksrc = ckv + (rowbase + j * 64 + r) * L_ + l0 + c;
#if USE_ASYNC_LDS
#pragma unroll
      for (int i = 0; i < 4; ++i) g2l_b128(qsrc + i * 8, &Qs[r][c] + i * 8);
#pragma unroll
      for (int i = 0; i < 4; ++i) g2l_b128(ksrc + i * 8, &Ks[r][c] + i * 8);
      __builtin_amdgcn_s_wait_asynccnt(0);
#else
      {
        const v8bf* qs8 = (const v8bf*)qsrc;
        const v8bf* ks8 = (const v8bf*)ksrc;
        v8bf* qd = (v8bf*)&Qs[r][c];
        v8bf* kd = (v8bf*)&Ks[r][c];
#pragma unroll
        for (int i = 0; i < 4; ++i) qd[i] = qs8[i];
#pragma unroll
        for (int i = 0; i < 4; ++i) kd[i] = ks8[i];
      }
#endif
      __syncthreads();

#pragma unroll
      for (int kk = 0; kk < 64; kk += 32) {
        const v16bf aq = load_frag_a_rm(&Qs[wave * 16][kk], 72);
#pragma unroll
        for (int t = 0; t < 4; ++t) {
          const v16bf bk = load_frag_b_t(&Ks[t * 16][kk], 72);   // B(k=l, n=key)
          sacc[t] = wmma_bf16(aq, bk, sacc[t]);
        }
      }
      __syncthreads();
    }

    // ---- scale, causal mask, online softmax update ----
#pragma unroll
    for (int v = 0; v < 8; ++v) {
      const int qrow = qr0 + wave * 16 + mb + v;
      float mx = -3.0e38f;
#pragma unroll
      for (int t = 0; t < 4; ++t) {
        const int key = j * 64 + t * 16 + nloc;
        float x = sacc[t][v] * 0.125f;          // 1/sqrt(DH)
        x = (key <= qrow) ? x : -3.0e38f;
        sacc[t][v] = x;
        mx = fmaxf(mx, x);
      }
#pragma unroll
      for (int d = 1; d < 16; d <<= 1) mx = fmaxf(mx, __shfl_xor(mx, d, 32));
      const float mnew  = fmaxf(mrow[v], mx);
      const float alpha = __expf(mrow[v] - mnew);
      float rsum = 0.f;
#pragma unroll
      for (int t = 0; t < 4; ++t) {
        const float p = __expf(sacc[t][v] - mnew);
        sacc[t][v] = p;
        rsum += p;
      }
#pragma unroll
      for (int d = 1; d < 16; d <<= 1) rsum += __shfl_xor(rsum, d, 32);
      mrow[v] = mnew;
      lrow[v] = lrow[v] * alpha + rsum;
#pragma unroll
      for (int t = 0; t < 4; ++t) cacc[t][v] *= alpha;
    }

    // ---- P -> LDS (bf16), A-fragment (row-major) layout ----
#pragma unroll
    for (int t = 0; t < 4; ++t)
#pragma unroll
      for (int v = 0; v < 8; ++v)
        Ps[wave * 16 + mb + v][t * 16 + nloc] = (__bf16)sacc[t][v];
    __syncthreads();

    // ---- stage V tile transposed [dh][key] from vmat_t [D][BS] ----
    {
      const int r = tid >> 1;            // dh 0..63
      const int c = (tid & 1) * 32;      // key chunk
      const __bf16* vsrc = vmat_t + (long)(h * DH_ + r) * BS_ + rowbase + j * 64 + c;
#if USE_ASYNC_LDS
#pragma unroll
      for (int i = 0; i < 4; ++i) g2l_b128(vsrc + i * 8, &VsT[r][c] + i * 8);
      __builtin_amdgcn_s_wait_asynccnt(0);
#else
      const v8bf* vs8 = (const v8bf*)vsrc;
      v8bf* vd = (v8bf*)&VsT[r][c];
#pragma unroll
      for (int i = 0; i < 4; ++i) vd[i] = vs8[i];
#endif
    }
    __syncthreads();

    // ---- ctx += P[64,64] @ V[64,64] ----
#pragma unroll
    for (int kk = 0; kk < 64; kk += 32) {
      const v16bf ap = load_frag_a_rm(&Ps[wave * 16][kk], 72);
#pragma unroll
      for (int t = 0; t < 4; ++t) {
        const v16bf bv = load_frag_b_t(&VsT[t * 16][kk], 72);    // B(k=key, n=dh)
        cacc[t] = wmma_bf16(ap, bv, cacc[t]);
      }
    }
    __syncthreads();
  }

  // ---- finalize: divide by softmax denominator, scatter to [BS][D] ----
#pragma unroll
  for (int t = 0; t < 4; ++t) {
#pragma unroll
    for (int v = 0; v < 8; ++v) {
      const long qrow = rowbase + qr0 + wave * 16 + mb + v;
      ctx[qrow * D_ + h * DH_ + t * 16 + nloc] = cacc[t][v] / lrow[v];
    }
  }
}

// ---------------- host-side orchestration ----------------
extern "C" void kernel_launch(void* const* d_in, const int* in_sizes, int n_in,
                              void* d_out, int out_size, void* d_ws, size_t ws_size,
                              hipStream_t stream) {
  (void)in_sizes; (void)n_in; (void)out_size; (void)ws_size;

  const float* x      = (const float*)d_in[0];   // [B,S,D]
  const float* W_q    = (const float*)d_in[1];   // [D,D]
  const float* W_dkv  = (const float*)d_in[2];   // [L,D]
  const float* W_uk   = (const float*)d_in[3];   // [D,L]
  const float* W_uv   = (const float*)d_in[4];   // [D,L]
  const float* W_o    = (const float*)d_in[5];   // [D,D]
  const float* gamma  = (const float*)d_in[6];   // [L]
  const float* beta   = (const float*)d_in[7];   // [L]

  float* out_f   = (float*)d_out;                       // [B,S,D]
  float* ckv_f   = (float*)d_out + (size_t)BS_ * D_;    // [B,S,L]

  // workspace carve-up
  char* ws = (char*)d_ws;
  float*  c_pre   = (float*)(ws);                                       // BS*L f32   (8 MiB)
  float*  ak_f32  = (float*)(ws + (size_t)BS_ * L_ * 4);                // D*L  f32   (2 MiB)
  __bf16* ckv_bf  = (__bf16*)(ws + (size_t)BS_ * L_ * 4
                                 + (size_t)D_  * L_ * 4);               // BS*L bf16  (4 MiB)
  __bf16* v_bf_t  = (__bf16*)((char*)ckv_bf + (size_t)BS_ * L_ * 2);    // D*BS bf16  (8 MiB), transposed
  float*  ctx_f   = (float*) ((char*)v_bf_t + (size_t)BS_ * D_ * 2);    // BS*D f32   (16 MiB)
  __bf16* tmp_bf  = (__bf16*)((char*)ctx_f  + (size_t)BS_ * D_ * 4);    // H*BS*L bf16 (64 MiB)

  // 1) absorbed_k (f32) = W_q[1024,1024] @ W_uk[1024,512]
  gemm_bf16_wmma<<<dim3(L_ / TN, D_ / TM, 1), 128, 0, stream>>>(
      W_q, D_, 0, W_uk, L_, 0, /*transB=*/0,
      ak_f32, nullptr, L_, 0, /*transC=*/0, D_, L_, D_);

  // 2) c = x[4096,1024] @ W_dkv^T (W_dkv stored [L,D])
  gemm_bf16_wmma<<<dim3(L_ / TN, BS_ / TM, 1), 128, 0, stream>>>(
      x, D_, 0, W_dkv, D_, 0, /*transB=*/1,
      c_pre, nullptr, L_, 0, /*transC=*/0, BS_, L_, D_);

  // 3) c_kv = LayerNorm(c): f32 to d_out (2nd output) + bf16 to ws
  layernorm_kernel<<<dim3(BS_, 1, 1), 256, 0, stream>>>(
      c_pre, gamma, beta, ckv_f, ckv_bf);

  // 4) tmp[h] = x[:, h*64:(h+1)*64] @ absorbed_k[h*64:(h+1)*64, :]  -> [H][BS][L] bf16
  gemm_bf16_wmma<<<dim3(L_ / TN, BS_ / TM, H_), 128, 0, stream>>>(
      x, D_, /*sAz=*/DH_, ak_f32, L_, /*sBz=*/(long)DH_ * L_, /*transB=*/0,
      nullptr, tmp_bf, L_, /*sCz=*/(long)BS_ * L_, /*transC=*/0, BS_, L_, DH_);

  // 5) v^T = (c_kv @ W_uv^T)^T -> [D][BS] bf16 (transC epilogue)
  gemm_bf16_wmma<<<dim3(D_ / TN, BS_ / TM, 1), 128, 0, stream>>>(
      ckv_f, L_, 0, W_uv, L_, 0, /*transB=*/1,
      nullptr, v_bf_t, /*ldc=*/BS_, 0, /*transC=*/1, BS_, D_, L_);

  // 6) flash attention: ctx[BS][D] f32
  mla_flash_kernel<<<dim3(S_ / 64, H_, B_), 128, 0, stream>>>(
      tmp_bf, ckv_bf, v_bf_t, ctx_f);

  // 7) out = ctx[4096,1024] @ W_o^T (W_o stored [D,D]) -> d_out (1st output)
  gemm_bf16_wmma<<<dim3(D_ / TN, BS_ / TM, 1), 128, 0, stream>>>(
      ctx_f, D_, 0, W_o, D_, 0, /*transB=*/1,
      out_f, nullptr, D_, 0, /*transC=*/0, BS_, D_, D_);
}
